// L_ECout_88648124991343
// MI455X (gfx1250) — matrix operand order repaired and checked by reading, not compiled, not run on hardware
//
#include <hip/hip_runtime.h>
#include <hip/hip_bf16.h>
#include <stdint.h>
#include <stddef.h>

// ---------------------------------------------------------------------------
// Problem constants (match reference)
// ---------------------------------------------------------------------------
#define M_DIM   256      // batch B
#define K_DIM   8192     // N_CA1 (reduction dim)
#define N_DIM   8192     // N_ITEMS
#define TOPK    819      // int(0.1 * 8192)
#define TAU     0.1f

// GEMM tiling
#define TILE_N  64       // columns per workgroup
#define KC      64       // K chunk per stage = 2 bf16-WMMA k-steps
#define PITCH_A 72       // LDS pitch (bf16): 144B -> 16B aligned, 36-dword bank stride
#define PITCH_B 72

typedef __attribute__((ext_vector_type(16))) __bf16        v16bf;
typedef __attribute__((ext_vector_type(2)))  __bf16        bf16x2;
typedef __attribute__((ext_vector_type(8)))  float         v8f;
typedef __attribute__((ext_vector_type(4)))  float         f32x4;
typedef __attribute__((ext_vector_type(4)))  unsigned int  u32x4;
typedef __attribute__((ext_vector_type(8)))  int           i32x8;
typedef __attribute__((ext_vector_type(4)))  int           i32x4;

// Native bf16 conversion (RNE) + pair pack -> v_cvt_pk_bf16_f32.
static __device__ __forceinline__ unsigned int pack_bf16(float lo, float hi) {
  bf16x2 p = { (__bf16)lo, (__bf16)hi };
  return __builtin_bit_cast(unsigned int, p);
}

static __device__ __forceinline__ float fast_rcp(float x) {
#if __has_builtin(__builtin_amdgcn_rcpf)
  return __builtin_amdgcn_rcpf(x);
#else
  return 1.0f / x;
#endif
}

// ---------------------------------------------------------------------------
// Kernel 1: convert a_CA1 (f32) -> bf16 once
// ---------------------------------------------------------------------------
__global__ __launch_bounds__(256) void cvt_a_kernel(const float* __restrict__ a,
                                                    unsigned short* __restrict__ abf) {
  int i = (blockIdx.x * 256 + threadIdx.x) * 4;   // M_DIM*K_DIM divisible by 4*256
  f32x4 v = *(const f32x4*)(a + i);
  *(uint2*)(abf + i) = make_uint2(pack_bf16(v[0], v[1]), pack_bf16(v[2], v[3]));
}

// ---------------------------------------------------------------------------
// Tensor Data Mover path for the A tile
// ---------------------------------------------------------------------------
#if __has_builtin(__builtin_amdgcn_tensor_load_to_lds) && \
    __has_builtin(__builtin_amdgcn_s_wait_tensorcnt)
#define HAVE_TDM 1
#else
#define HAVE_TDM 0
#endif

#if HAVE_TDM
// Load A tile (256 rows x KC=64 bf16 cols at column k0) into LDS at lds_off.
// TDM padding: pad_interval=32 dwords (one 64-elem bf16 row = 128B),
// pad_amount=4 dwords (16B) -> effective LDS pitch = 72 bf16 = PITCH_A.
static __device__ __forceinline__ void tdm_load_A(const unsigned short* __restrict__ Abf,
                                                  int k0, unsigned int lds_off) {
  unsigned long long ga = (unsigned long long)(uintptr_t)Abf + (unsigned long long)k0 * 2ull;
  u32x4 g0;
  g0[0] = 1u;                                              // count=1, no gather
  g0[1] = lds_off;                                         // lds_addr (bytes)
  g0[2] = (unsigned int)ga;                                // global_addr[31:0]
  g0[3] = (unsigned int)((ga >> 32) & 0x01FFFFFFull)       // global_addr[56:32]
          | (2u << 30);                                    // type = 2 ("image")
  i32x8 g1;
  g1[0] = (int)((1u << 16)      // data_size = 1 -> 2 bytes
              | (1u << 20)      // pad_enable
              | (4u << 22)      // pad_interval: 32 dwords
              | (3u << 25));    // pad_amount: 4 dwords
  g1[1] = (int)(8192u << 16);   // tensor_dim0 = 8192 (low 16 bits)
  g1[2] = (int)(256u  << 16);   // tensor_dim0 hi | tensor_dim1 = 256
  g1[3] = (int)(64u   << 16);   // tensor_dim1 hi | tile_dim0 = 64
  g1[4] = (int)(256u);          // tile_dim1 = 256, tile_dim2 = 0
  g1[5] = (int)(8192u);         // tensor_dim0_stride lo = 8192 elems
  g1[6] = 0;
  g1[7] = 0;
  i32x4 gz4 = {0, 0, 0, 0};     // 2D: groups 2/3 unused
  i32x8 gz8 = {0, 0, 0, 0, 0, 0, 0, 0};
  __builtin_amdgcn_tensor_load_to_lds(g0, g1, gz4, gz4, gz8, 0);  // 6-arg form
}
#endif

// ---------------------------------------------------------------------------
// Kernel 2: bf16 WMMA GEMM (full M strip per WG) + relu/(relu+1) epilogue
// ---------------------------------------------------------------------------
struct WStage { f32x4 w0[2], w1[2]; };    // 2 chunks x (two k-adjacent W rows)

static __device__ __forceinline__ void wstage_load(WStage& st, const float* __restrict__ W,
                                                   int k0, int n0, int tid) {
#pragma unroll
  for (int q = 0; q < 2; ++q) {
    const int id = tid + q * 256;         // 0..511
    const int kk = (id >> 4) * 2;         // k pair {kk, kk+1}, kk in 0..62
    const int c4 = (id & 15) << 2;        // n offset 0..60
    // W is single-use: non-temporal so it doesn't evict A / act from L2
    st.w0[q] = __builtin_nontemporal_load((const f32x4*)(W + (size_t)(k0 + kk)     * N_DIM + n0 + c4));
    st.w1[q] = __builtin_nontemporal_load((const f32x4*)(W + (size_t)(k0 + kk + 1) * N_DIM + n0 + c4));
  }
}

static __device__ __forceinline__ void wstage_store(const WStage& st,
                                                    unsigned short* __restrict__ sB, int tid) {
#pragma unroll
  for (int q = 0; q < 2; ++q) {
    const int id = tid + q * 256;
    const int kk = (id >> 4) * 2;
    const int c4 = (id & 15) << 2;
#pragma unroll
    for (int j = 0; j < 4; ++j)           // [n][k],[n][k+1] adjacent -> one b32 store
      *(unsigned int*)(sB + (c4 + j) * PITCH_B + kk) = pack_bf16(st.w0[q][j], st.w1[q][j]);
  }
}

union FragU { u32x4 u[2]; v16bf v; };

// One KC=64 tile of WMMAs: 2 k-steps x (2 m x 4 n) accumulations.
static __device__ __forceinline__ void compute_tile(const unsigned short* __restrict__ sAcur,
                                                    const unsigned short* __restrict__ sB,
                                                    v8f acc[2][4], int mrow, int lane) {
  const int lrow  = lane & 15;
  const int khalf = (lane < 16) ? 0 : 8;  // A: lanes>=16 hold K 8..15 / 24..31
  const int kb    = (lane < 16) ? 0 : 8;  // B: lanes>=16 hold K 8..15 / 24..31
#pragma unroll
  for (int ks = 0; ks < 2; ++ks) {        // k-step: columns ks*32 .. ks*32+31
    FragU afrag[2], bfrag[4];
#pragma unroll
    for (int m = 0; m < 2; ++m) {
      const unsigned short* ap = sAcur + (mrow + m * 16 + lrow) * PITCH_A + ks * 32 + khalf;
      afrag[m].u[0] = *(const u32x4*)(ap);        // K 0..7  (or 8..15)
      afrag[m].u[1] = *(const u32x4*)(ap + 16);   // K 16..23 (or 24..31)
    }
#pragma unroll
    for (int nj = 0; nj < 4; ++nj) {
      const unsigned short* bp = sB + (nj * 16 + lrow) * PITCH_B + ks * 32 + kb;
      bfrag[nj].u[0] = *(const u32x4*)(bp);       // K 0..7  (or 8..15)
      bfrag[nj].u[1] = *(const u32x4*)(bp + 16);  // K 16..23 (or 24..31)
    }
#pragma unroll
    for (int m = 0; m < 2; ++m)
#pragma unroll
      for (int nj = 0; nj < 4; ++nj)
        acc[m][nj] = __builtin_amdgcn_wmma_f32_16x16x32_bf16(
            false, afrag[m].v, false, bfrag[nj].v,
            (short)0, acc[m][nj], false, false);
  }
}

// One pipeline round: consume sAcur (tile at k0), prefetch k0+KC into the
// other buffer via TDM, stage W through registers -> LDS.
static __device__ __forceinline__ void gemm_round(const unsigned short* __restrict__ Abf,
                                                  const float* __restrict__ W,
                                                  const unsigned short* __restrict__ sAcur,
                                                  unsigned int ldsoff_nxt,
                                                  unsigned short* __restrict__ sB,
                                                  WStage& ws, v8f acc[2][4],
                                                  int k0, int n0, int tid, int wave,
                                                  int lane, int mrow) {
  __syncthreads();                        // prior round's sB/sA[next] readers done
  wstage_store(ws, sB, tid);
#if HAVE_TDM
  if (wave == 0) {
    if (k0 + KC < K_DIM) {
      tdm_load_A(Abf, k0 + KC, ldsoff_nxt);
      __builtin_amdgcn_s_wait_tensorcnt(1);   // this round's tile done; next in flight
    } else {
      __builtin_amdgcn_s_wait_tensorcnt(0);
    }
  }
#else
  // Fallback: direct copy of the current tile (no pipelining).
  {
    unsigned short* sAw = (unsigned short*)sAcur;
#pragma unroll
    for (int j = 0; j < 8; ++j) {
      const int id  = tid + j * 256;      // 0..2047 chunks of 16B
      const int row = id >> 3;
      const int kc  = (id & 7) * 8;
      *(u32x4*)(sAw + row * PITCH_A + kc) =
          *(const u32x4*)(Abf + (size_t)row * K_DIM + k0 + kc);
    }
  }
#endif
  __syncthreads();                        // tile + sB visible to all waves
  if (k0 + KC < K_DIM)
    wstage_load(ws, W, k0 + KC, n0, tid); // overlap next global loads with WMMA
  compute_tile(sAcur, sB, acc, mrow, lane);
}

__global__ __launch_bounds__(256, 2) void gemm_act_kernel(
    const unsigned short* __restrict__ Abf,   // 256 x 8192 bf16
    const float* __restrict__ W,              // 8192 x 8192 f32
    float* __restrict__ act)                  // 256 x 8192 f32
{
  __shared__ __align__(16) unsigned short sA0[M_DIM * PITCH_A];   // 36864 B
  __shared__ __align__(16) unsigned short sA1[M_DIM * PITCH_A];   // 36864 B
  __shared__ __align__(16) unsigned short sB[TILE_N * PITCH_B];   //  9216 B

  const int tid  = threadIdx.x;
  const int wave = tid >> 5;
  const int lane = tid & 31;
  const int n0   = blockIdx.x * TILE_N;
  const int mrow = wave * 32;               // this wave's 32-row strip

  v8f acc[2][4];
#pragma unroll
  for (int m = 0; m < 2; ++m)
#pragma unroll
    for (int nj = 0; nj < 4; ++nj)
      acc[m][nj] = (v8f){0.f, 0.f, 0.f, 0.f, 0.f, 0.f, 0.f, 0.f};

  const unsigned int off0 = (unsigned int)(uintptr_t)(void*)&sA0[0];
  const unsigned int off1 = (unsigned int)(uintptr_t)(void*)&sA1[0];

  WStage ws;
  wstage_load(ws, W, 0, n0, tid);
#if HAVE_TDM
  if (wave == 0) tdm_load_A(Abf, 0, off0);
#endif

  // Manual unroll by 2: compile-time-constant LDS bases per half, no parity
  // branch -> accumulators stay register-resident (D tied to C).
  for (int kt = 0; kt < K_DIM / KC; kt += 2) {
    gemm_round(Abf, W, sA0, off1, sB, ws, acc, (kt + 0) * KC, n0, tid, wave, lane, mrow);
    gemm_round(Abf, W, sA1, off0, sB, ws, acc, (kt + 1) * KC, n0, tid, wave, lane, mrow);
  }

  // Epilogue: act = relu(net) / (relu(net)+1) via v_rcp_f32.
  // C layout: VGPR v -> (lanes 0-15: M=v, N=lane) (lanes 16-31: M=v+8, N=lane-16)
  const int lrow  = lane & 15;
  const int rhalf = (lane < 16) ? 0 : 8;
#pragma unroll
  for (int m = 0; m < 2; ++m) {
    const int rbase = mrow + m * 16 + rhalf;
#pragma unroll
    for (int nj = 0; nj < 4; ++nj) {
      const int col = n0 + nj * 16 + lrow;
#pragma unroll
      for (int v = 0; v < 8; ++v) {
        float pos = fmaxf(acc[m][nj][v], 0.0f);
        act[(size_t)(rbase + v) * N_DIM + col] = pos * fast_rcp(pos + 1.0f);
      }
    }
  }
}

// ---------------------------------------------------------------------------
// Kernel 3: exact per-row K-th largest via 4-pass radix-256 select on float
// bits (all act >= 0, so uint order == float order). One WG per row.
// ---------------------------------------------------------------------------
__global__ __launch_bounds__(256) void topk_thresh_kernel(
    const float* __restrict__ act, float* __restrict__ thr) {
  __shared__ unsigned int hist[256];
  __shared__ unsigned int s_prefix;
  __shared__ unsigned int s_remaining;

  const int row = blockIdx.x;
  const float* arow = act + (size_t)row * N_DIM;
  if (threadIdx.x == 0) { s_prefix = 0u; s_remaining = TOPK; }
  __syncthreads();

  for (int b = 3; b >= 0; --b) {
    hist[threadIdx.x] = 0u;
    __syncthreads();
    const unsigned int prefix = s_prefix;
    const unsigned int himask = (b == 3) ? 0u : (0xFFFFFFFFu << ((b + 1) * 8));
    for (int it = 0; it < N_DIM / (256 * 4); ++it) {
      f32x4 vv = *(const f32x4*)(arow + (it * 256 + threadIdx.x) * 4);
#pragma unroll
      for (int j = 0; j < 4; ++j) {
        unsigned int u = __builtin_bit_cast(unsigned int, vv[j]);
        if ((u & himask) == (prefix & himask))
          atomicAdd(&hist[(u >> (b * 8)) & 0xFFu], 1u);
      }
    }
    __syncthreads();
    if (threadIdx.x == 0) {
      unsigned int rem = s_remaining, cum = 0u; int sel = 0;
      for (int d = 255; d >= 0; --d) {
        unsigned int c = hist[d];
        if (cum + c >= rem) { sel = d; s_remaining = rem - cum; break; }
        cum += c;
      }
      s_prefix = prefix | ((unsigned int)sel << (b * 8));
    }
    __syncthreads();
  }
  if (threadIdx.x == 0) thr[row] = __uint_as_float(s_prefix);
}

// ---------------------------------------------------------------------------
// Kernel 4: kWTA mask + EMA blend, vectorized x4
// ---------------------------------------------------------------------------
__global__ __launch_bounds__(256) void kwta_blend_kernel(
    const float* __restrict__ act, const float* __restrict__ activity,
    const float* __restrict__ thr, float* __restrict__ out) {
  const int i  = (blockIdx.x * 256 + threadIdx.x) * 4;  // N_DIM % 4 == 0
  const int row = i / N_DIM;
  const float t = thr[row];
  f32x4 a = *(const f32x4*)(act + i);
  f32x4 v = *(const f32x4*)(activity + i);
  f32x4 o;
#pragma unroll
  for (int j = 0; j < 4; ++j) {
    float kw = (a[j] >= t) ? a[j] : 0.0f;
    o[j] = v[j] + TAU * (kw - v[j]);
  }
  *(f32x4*)(out + i) = o;
}

// ---------------------------------------------------------------------------
// Launcher. Inputs: d_in[0]=a_CA1 (256x8192 f32), d_in[1]=W (8192x8192 f32),
// d_in[2]=activity (256x8192 f32). Output: 256x8192 f32.
// Workspace layout: [abf16: 4MB][act f32: 8MB][thr: 1KB] (~12.6 MB total)
// ---------------------------------------------------------------------------
extern "C" void kernel_launch(void* const* d_in, const int* in_sizes, int n_in,
                              void* d_out, int out_size, void* d_ws, size_t ws_size,
                              hipStream_t stream) {
  const float* a_CA1    = (const float*)d_in[0];
  const float* W        = (const float*)d_in[1];
  const float* activity = (const float*)d_in[2];
  float* out = (float*)d_out;

  char* ws = (char*)d_ws;
  unsigned short* abf = (unsigned short*)ws;                               // 256*8192*2
  float* actbuf = (float*)(ws + (size_t)M_DIM * K_DIM * sizeof(unsigned short));
  float* thr    = (float*)(ws + (size_t)M_DIM * K_DIM * sizeof(unsigned short)
                              + (size_t)M_DIM * N_DIM * sizeof(float));

  // 1) a_CA1 -> bf16
  cvt_a_kernel<<<(M_DIM * K_DIM) / (256 * 4), 256, 0, stream>>>(a_CA1, abf);
  // 2) bf16 WMMA GEMM + activation (A tile via TDM, double-buffered)
  gemm_act_kernel<<<N_DIM / TILE_N, 256, 0, stream>>>(abf, W, actbuf);
  // 3) exact per-row top-K threshold
  topk_thresh_kernel<<<M_DIM, 256, 0, stream>>>(actbuf, thr);
  // 4) kWTA + EMA blend
  kwta_blend_kernel<<<(M_DIM * N_DIM) / (256 * 4), 256, 0, stream>>>(actbuf, activity, thr, out);
}
